// GPT_24446953849213
// MI455X (gfx1250) — compile-verified, matrix operand order checked by reading, not run on hardware
//
#include <hip/hip_runtime.h>
#include <math.h>

// ---------------------------------------------------------------------------
// GPT forward for MI455X (gfx1250, wave32).
//  - All matmuls: v_wmma_f32_16x16x32_bf16 (f32 accumulate)
//  - Weights + GEMM activations pre-converted to bf16 in workspace
//  - GEMM tiles staged to LDS with global_load_async_to_lds_b128 (ASYNCcnt),
//    double-buffered so fill(k+1) overlaps WMMA(k)
//  - Attention V tiles staged by the Tensor Data Mover (tensor_load_to_lds,
//    TENSORcnt), one 2-D descriptor per 32-key chunk per wave
// ---------------------------------------------------------------------------

typedef __attribute__((ext_vector_type(16))) __bf16 v16bf;
typedef __attribute__((ext_vector_type(8)))  __bf16 v8bf;
typedef __attribute__((ext_vector_type(8)))  float  v8f;
typedef __attribute__((ext_vector_type(4)))  unsigned v4u;
typedef __attribute__((ext_vector_type(4)))  int    v4i;
typedef __attribute__((ext_vector_type(8)))  int    v8i;

#define TT   2048
#define DD   768
#define HH   12
#define LL   4
#define VV   32000
#define MM   4096          // B*T
#define HDIM 64

__device__ __forceinline__ v8f wmma_bf16(v16bf a, v16bf b, v8f c) {
  return __builtin_amdgcn_wmma_f32_16x16x32_bf16(false, a, false, b, (short)0, c,
                                                 false, false);
}

__device__ __forceinline__ float gelu_exact(float v) {
  return 0.5f * v * (1.0f + erff(v * 0.70710678118654752f));
}

// Async DMA 16B global -> LDS (tracked by ASYNCcnt, not via VGPRs).
__device__ __forceinline__ void async_b128(unsigned lds_off, const __bf16* g) {
  unsigned long long ga = (unsigned long long)(size_t)g;
  asm volatile("global_load_async_to_lds_b128 %0, %1, off"
               :: "v"(lds_off), "v"(ga) : "memory");
}
__device__ __forceinline__ void wait_async0() {
  asm volatile("s_wait_asynccnt 0x0" ::: "memory");
}
// LDS byte offset of a __shared__ object = low 32 bits of its flat address.
__device__ __forceinline__ unsigned lds_off_of(const void* p) {
  return (unsigned)(unsigned long long)(size_t)p;
}

// Tensor Data Mover: DMA a 2-D tile (tile_d0 x tile_d1 of 2-byte elements,
// row stride in elements) from global into LDS at lds_addr. D# per CDNA5 ISA
// ch.8: group0 = {count, lds_addr, global_addr[56:0], type=2},
// group1 = {data_size=2B, tensor dims (oversized: no OOB clip), tile dims,
// dim0 stride}. Groups 2/3 zero (2-D tile).
__device__ __forceinline__ void tdm_load_2d(unsigned lds_addr, const void* gp,
                                            int tile_d0, int tile_d1,
                                            int row_stride_elems) {
  unsigned long long ga = (unsigned long long)(size_t)gp;
  v4u g0;
  g0[0] = 1u;                                         // count=1 (valid), user
  g0[1] = lds_addr;                                   // LDS byte address
  g0[2] = (unsigned)ga;                               // global_addr[31:0]
  g0[3] = (unsigned)((ga >> 32) & 0x01FFFFFFull)      // global_addr[56:32]
          | 0x80000000u;                              // type=2 ("image")
  const unsigned TD = 0x00100000u;                    // huge tensor dims
  v8i g1;
  g1[0] = (int)(1u << 16);                            // data_size = 2 bytes
  g1[1] = (int)((TD & 0xFFFFu) << 16);                // tensor_dim0[15:0]
  g1[2] = (int)(((TD >> 16) & 0xFFFFu)                // tensor_dim0[31:16]
          | ((TD & 0xFFFFu) << 16));                  // tensor_dim1[15:0]
  g1[3] = (int)(((TD >> 16) & 0xFFFFu)                // tensor_dim1[31:16]
          | ((unsigned)tile_d0 << 16));               // tile_dim0
  g1[4] = tile_d1;                                    // tile_dim1 (dim2=0)
  g1[5] = row_stride_elems;                           // dim0_stride[31:0]
  g1[6] = 0;
  g1[7] = 0;
  v4i gz = {0, 0, 0, 0};
#if defined(__clang_major__) && __clang_major__ >= 23
  v8i gz8 = {0, 0, 0, 0, 0, 0, 0, 0};
  __builtin_amdgcn_tensor_load_to_lds(g0, g1, gz, gz, gz8, 0);
#else
  __builtin_amdgcn_tensor_load_to_lds(g0, g1, gz, gz, 0);
#endif
}

// 16x32 bf16 fragment (A-row-layout / B-col-layout, symmetric assumption):
// lane L -> row/col (L&15); elems 0..7 <-> K=b0+0..7, 8..15 <-> K=b0+16..23,
// b0 = (L<16)?0:8.  Works for LDS or global bf16 pointers (v8bf loads).
__device__ __forceinline__ v16bf ldfrag_bf(const __bf16* rowbase, int b0) {
  v8bf lo = *(const v8bf*)(rowbase + b0);
  v8bf hi = *(const v8bf*)(rowbase + b0 + 16);
  v16bf v;
#pragma unroll
  for (int i = 0; i < 8; ++i) { v[i] = lo[i]; v[i + 8] = hi[i]; }
  return v;
}
// Same pattern from f32 source with conversion (used for the P tile).
__device__ __forceinline__ v16bf ldfrag_f32(const float* rowbase, int b0) {
  v16bf v;
#pragma unroll
  for (int i = 0; i < 8; ++i) {
    v[i]     = (__bf16)rowbase[b0 + i];
    v[i + 8] = (__bf16)rowbase[b0 + 16 + i];
  }
  return v;
}

// ---------------------------------------------------------------------------
// f32 -> bf16 bulk convert (weights once per launch; bandwidth-trivial)
// ---------------------------------------------------------------------------
__global__ __launch_bounds__(256) void cvt_kernel(const float* __restrict__ in,
                                                  __bf16* __restrict__ out,
                                                  int n) {
  int stride = gridDim.x * 256;
  for (int i = blockIdx.x * 256 + threadIdx.x; i < n; i += stride)
    out[i] = (__bf16)in[i];
}

// ---------------------------------------------------------------------------
// Embedding: x[b,t,:] = tok_emb[id] + pos_emb[t]   (residual stream is f32)
// ---------------------------------------------------------------------------
__global__ __launch_bounds__(256) void embed_kernel(
    const int* __restrict__ ids, const float* __restrict__ tok,
    const float* __restrict__ pos, float* __restrict__ x) {
  int row = blockIdx.x;
  int t   = row & (TT - 1);
  int id  = ids[row];
  size_t xb = (size_t)row * DD, tb = (size_t)id * DD, pb = (size_t)t * DD;
#pragma unroll
  for (int i = 0; i < 3; ++i) {
    int c = threadIdx.x + 256 * i;
    x[xb + c] = tok[tb + c] + pos[pb + c];
  }
}

// ---------------------------------------------------------------------------
// LayerNorm: f32 in, bf16 out (feeds WMMA GEMMs)
// ---------------------------------------------------------------------------
__global__ __launch_bounds__(256) void ln_kernel(
    const float* __restrict__ x, const float* __restrict__ g,
    const float* __restrict__ b, __bf16* __restrict__ out) {
  __shared__ float red[256];
  int tid = threadIdx.x;
  size_t base = (size_t)blockIdx.x * DD;
  float v0 = x[base + tid];
  float v1 = x[base + tid + 256];
  float v2 = x[base + tid + 512];
  red[tid] = v0 + v1 + v2;
  __syncthreads();
  for (int o = 128; o > 0; o >>= 1) {
    if (tid < o) red[tid] += red[tid + o];
    __syncthreads();
  }
  float mu = red[0] * (1.0f / DD);
  __syncthreads();
  float d0 = v0 - mu, d1 = v1 - mu, d2 = v2 - mu;
  red[tid] = d0 * d0 + d1 * d1 + d2 * d2;
  __syncthreads();
  for (int o = 128; o > 0; o >>= 1) {
    if (tid < o) red[tid] += red[tid + o];
    __syncthreads();
  }
  float rstd = rsqrtf(red[0] * (1.0f / DD) + 1e-5f);
  out[base + tid]       = (__bf16)(d0 * rstd * g[tid]       + b[tid]);
  out[base + tid + 256] = (__bf16)(d1 * rstd * g[tid + 256] + b[tid + 256]);
  out[base + tid + 512] = (__bf16)(d2 * rstd * g[tid + 512] + b[tid + 512]);
}

// ---------------------------------------------------------------------------
// GEMM: out = act(A[M,K](bf16) @ W[N,K](bf16)^T + bias) (+res f32)
// Block 128x128x32, 8 waves (wave = 64x32). Tiles DMA'd to LDS via
// global_load_async_to_lds_b128, double-buffered.
// outf!=null -> f32 store (+optional res); else bf16 store to outb.
// ---------------------------------------------------------------------------
#define BM 128
#define BN 128
#define BK 32

__global__ __launch_bounds__(256) void gemm_kernel(
    const __bf16* __restrict__ A, const __bf16* __restrict__ W,
    const float* __restrict__ bias, const float* __restrict__ res,
    float* __restrict__ outf, __bf16* __restrict__ outb,
    int N, int K, int act) {
  __shared__ __align__(16) __bf16 As[2][BM * BK];   // 8KB each
  __shared__ __align__(16) __bf16 Ws[2][BN * BK];

  int tid  = threadIdx.x;
  int lane = tid & 31;
  int w    = tid >> 5;
  int wm   = w >> 2;            // 0..1
  int wn   = w & 3;             // 0..3
  int l16  = lane & 15;
  int hf   = lane >> 4;
  int b0   = hf * 8;

  int bm = blockIdx.y * BM;
  int bn = blockIdx.x * BN;

  v8f acc[4][2];
  v8f z = {};
#pragma unroll
  for (int mt = 0; mt < 4; ++mt)
#pragma unroll
    for (int nt = 0; nt < 2; ++nt) acc[mt][nt] = z;

  // per-thread DMA slice: row = tid/4 (0..63, +64 for second half), 16B chunk
  int drow   = tid >> 2;
  int dchunk = tid & 3;
  unsigned as_base = lds_off_of(&As[0][0]) + drow * 64 + dchunk * 16;
  unsigned ws_base = lds_off_of(&Ws[0][0]) + drow * 64 + dchunk * 16;
  const __bf16* ga = A + (size_t)(bm + drow) * K + dchunk * 8;
  const __bf16* gw = W + (size_t)(bn + drow) * K + dchunk * 8;
  const size_t half_rows = (size_t)64 * K;

#define ISSUE_TILES(buf, kk)                                                   \
  do {                                                                         \
    unsigned lo = (unsigned)(buf) * (BM * BK * 2);                             \
    async_b128(as_base + lo,        ga + (kk));                                \
    async_b128(as_base + lo + 4096, ga + (kk) + half_rows);                    \
    async_b128(ws_base + lo,        gw + (kk));                                \
    async_b128(ws_base + lo + 4096, gw + (kk) + half_rows);                    \
  } while (0)

  ISSUE_TILES(0, 0);

  for (int kk = 0; kk < K; kk += BK) {
    int buf = (kk >> 5) & 1;
    wait_async0();        // this wave's fills for `buf` landed in LDS
    __syncthreads();      // everyone's fills landed; prior reads done
    if (kk + BK < K) ISSUE_TILES(buf ^ 1, kk + BK);

    v16bf af[4], wf[2];
#pragma unroll
    for (int mt = 0; mt < 4; ++mt)
      af[mt] = ldfrag_bf(&As[buf][(wm * 64 + mt * 16 + l16) * BK], b0);
#pragma unroll
    for (int nt = 0; nt < 2; ++nt)
      wf[nt] = ldfrag_bf(&Ws[buf][(wn * 32 + nt * 16 + l16) * BK], b0);
#pragma unroll
    for (int mt = 0; mt < 4; ++mt)
#pragma unroll
      for (int nt = 0; nt < 2; ++nt)
        acc[mt][nt] = wmma_bf16(af[mt], wf[nt], acc[mt][nt]);
    __syncthreads();
  }
#undef ISSUE_TILES

  // Epilogue. C layout: lane -> N = l16, VGPR r -> M = r + 8*half.
#pragma unroll
  for (int nt = 0; nt < 2; ++nt) {
    int col = bn + wn * 32 + nt * 16 + l16;
    float bv = bias ? bias[col] : 0.0f;
#pragma unroll
    for (int mt = 0; mt < 4; ++mt) {
#pragma unroll
      for (int r = 0; r < 8; ++r) {
        int row = bm + wm * 64 + mt * 16 + r + hf * 8;
        float v = acc[mt][nt][r] + bv;
        if (act == 1) v = gelu_exact(v);
        if (outf) {
          if (res) v += res[(size_t)row * N + col];
          outf[(size_t)row * N + col] = v;
        } else {
          outb[(size_t)row * N + col] = (__bf16)v;
        }
      }
    }
  }
}

// ---------------------------------------------------------------------------
// Causal flash attention. One wave per (b, head, 16-query tile).
// qkv is bf16 (B*T, 3D); q @ h*64, k @ D+h*64, v @ 2D+h*64. out: bf16 (B*T,D)
// V chunk (32x64 bf16 2-D tile) staged by the Tensor Data Mover, overlapped
// with the S = QK^T WMMAs; completion via TENSORcnt.
// ---------------------------------------------------------------------------
__global__ __launch_bounds__(32) void attn_kernel(
    const __bf16* __restrict__ qkv, __bf16* __restrict__ out) {
  __shared__ __align__(128) float  Pl[16 * 32];   // exp(S), C->A relayout
  __shared__ __align__(128) __bf16 Vl[32 * 64];   // staged V chunk (4KB)

  int lane = threadIdx.x & 31;
  int blk  = blockIdx.x;
  int qb   = blk & 127;               // T/16 = 128
  int h    = (blk >> 7) % HH;
  int b    = blk / (128 * HH);
  int qbase = qb * 16;
  int l16 = lane & 15;
  int hf  = lane >> 4;
  int b0  = hf * 8;
  const int LD = 3 * DD;
  const float scale = 0.125f;
  size_t tokbase = (size_t)b * TT;
  unsigned vl_addr = lds_off_of(&Vl[0]);

  v16bf qf[2];
  {
    const __bf16* qrow = qkv + (tokbase + qbase + l16) * LD + h * HDIM;
    qf[0] = ldfrag_bf(qrow, b0);
    qf[1] = ldfrag_bf(qrow + 32, b0);
  }

  v8f o[4];
  v8f z = {};
#pragma unroll
  for (int t = 0; t < 4; ++t) o[t] = z;
  float m[8], lsum[8];
#pragma unroll
  for (int r = 0; r < 8; ++r) { m[r] = -INFINITY; lsum[r] = 0.0f; }

  for (int j = 0; j <= qbase + 15; j += 32) {
    // TDM: stage V rows j..j+31 (32 x 64 bf16, row stride 3D) into LDS;
    // the DMA overlaps the S computation below.
    tdm_load_2d(vl_addr, qkv + (tokbase + j) * LD + 2 * DD + h * HDIM,
                HDIM, 32, LD);

    // S = Q @ K^T, 2 groups of 16 keys (K-dim 64 -> 2 wmma each)
    v8f s[2];
#pragma unroll
    for (int g = 0; g < 2; ++g) {
      const __bf16* krow =
          qkv + (tokbase + j + 16 * g + l16) * LD + DD + h * HDIM;
      v16bf kf0 = ldfrag_bf(krow, b0);
      v16bf kf1 = ldfrag_bf(krow + 32, b0);
      v8f sc = z;
      sc = wmma_bf16(qf[0], kf0, sc);
      sc = wmma_bf16(qf[1], kf1, sc);
      s[g] = sc;
    }

    // causal mask + scale; row max over 32 keys (xor-shuffles stay in half)
    float rmax[8];
#pragma unroll
    for (int r = 0; r < 8; ++r) {
      int qi = qbase + r + hf * 8;
      float s0 = ((j + l16)      <= qi) ? s[0][r] * scale : -1e30f;
      float s1 = ((j + 16 + l16) <= qi) ? s[1][r] * scale : -1e30f;
      s[0][r] = s0; s[1][r] = s1;
      float mx = fmaxf(s0, s1);
      for (int d = 1; d < 16; d <<= 1) mx = fmaxf(mx, __shfl_xor(mx, d, 32));
      rmax[r] = mx;
    }

    float alpha[8];
#pragma unroll
    for (int r = 0; r < 8; ++r) {
      float mn = fmaxf(m[r], rmax[r]);
      alpha[r] = __expf(m[r] - mn);   // first chunk: exp(-inf)=0
      m[r] = mn;
      float p0 = __expf(s[0][r] - mn);
      float p1 = __expf(s[1][r] - mn);
      s[0][r] = p0; s[1][r] = p1;
      float rs = p0 + p1;
      for (int d = 1; d < 16; d <<= 1) rs += __shfl_xor(rs, d, 32);
      lsum[r] = lsum[r] * alpha[r] + rs;
    }

    __builtin_amdgcn_s_wait_tensorcnt(0);   // V tile landed in LDS
    __syncthreads();

    // P (C layout) -> LDS -> A-fragment layout
#pragma unroll
    for (int r = 0; r < 8; ++r) {
      int row = r + hf * 8;
      Pl[row * 32 + l16]      = s[0][r];
      Pl[row * 32 + 16 + l16] = s[1][r];
    }
    __syncthreads();

#pragma unroll
    for (int t = 0; t < 4; ++t)
#pragma unroll
      for (int r = 0; r < 8; ++r) o[t][r] *= alpha[r];

    v16bf pf = ldfrag_f32(&Pl[l16 * 32], b0);

#pragma unroll
    for (int t = 0; t < 4; ++t) {
      v16bf vf;
#pragma unroll
      for (int i = 0; i < 8; ++i) {
        vf[i]     = Vl[(b0 + i) * 64 + 16 * t + l16];
        vf[i + 8] = Vl[(b0 + 16 + i) * 64 + 16 * t + l16];
      }
      o[t] = wmma_bf16(pf, vf, o[t]);
    }
    __syncthreads();     // reads done before next chunk's TDM fill
  }

  float inv[8];
#pragma unroll
  for (int r = 0; r < 8; ++r) inv[r] = 1.0f / lsum[r];
#pragma unroll
  for (int t = 0; t < 4; ++t) {
    int col = h * HDIM + 16 * t + l16;
#pragma unroll
    for (int r = 0; r < 8; ++r) {
      int row = qbase + r + hf * 8;
      out[(tokbase + row) * DD + col] = (__bf16)(o[t][r] * inv[r]);
    }
  }
}

// ---------------------------------------------------------------------------
// Host-side orchestration
// ---------------------------------------------------------------------------
extern "C" void kernel_launch(void* const* d_in, const int* in_sizes, int n_in,
                              void* d_out, int out_size, void* d_ws,
                              size_t ws_size, hipStream_t stream) {
  (void)in_sizes; (void)n_in; (void)out_size; (void)ws_size;

  const int*   ids   = (const int*)d_in[0];
  const float* tok   = (const float*)d_in[1];
  const float* pos   = (const float*)d_in[2];
  const float* ln1g  = (const float*)d_in[3];
  const float* ln1b  = (const float*)d_in[4];
  const float* Wqkv  = (const float*)d_in[5];
  const float* bqkv  = (const float*)d_in[6];
  const float* Wproj = (const float*)d_in[7];
  const float* bproj = (const float*)d_in[8];
  const float* ln2g  = (const float*)d_in[9];
  const float* ln2b  = (const float*)d_in[10];
  const float* W1    = (const float*)d_in[11];
  const float* b1    = (const float*)d_in[12];
  const float* W2    = (const float*)d_in[13];
  const float* b2    = (const float*)d_in[14];
  const float* lnfg  = (const float*)d_in[15];
  const float* lnfb  = (const float*)d_in[16];
  float* logits = (float*)d_out;

  // workspace carve-out (256B aligned chunks)
  char* wp = (char*)d_ws;
  auto carve = [&](size_t bytes) -> void* {
    void* p = (void*)wp;
    wp += (bytes + 255) & ~(size_t)255;
    return p;
  };
  float*  x     = (float*)carve((size_t)MM * DD * 4);
  __bf16* hb    = (__bf16*)carve((size_t)MM * DD * 2);
  __bf16* qkv   = (__bf16*)carve((size_t)MM * 3 * DD * 2);
  __bf16* att   = (__bf16*)carve((size_t)MM * DD * 2);
  __bf16* mid   = (__bf16*)carve((size_t)MM * 4 * DD * 2);
  __bf16* Wqkvb = (__bf16*)carve((size_t)LL * 3 * DD * DD * 2);
  __bf16* Wprob = (__bf16*)carve((size_t)LL * DD * DD * 2);
  __bf16* W1b   = (__bf16*)carve((size_t)LL * 4 * DD * DD * 2);
  __bf16* W2b   = (__bf16*)carve((size_t)LL * DD * 4 * DD * 2);
  __bf16* tokb  = (__bf16*)carve((size_t)VV * DD * 2);

  // one-time (per launch) bf16 weight conversion — bandwidth trivial
  cvt_kernel<<<4096, 256, 0, stream>>>(Wqkv,  Wqkvb, LL * 3 * DD * DD);
  cvt_kernel<<<2048, 256, 0, stream>>>(Wproj, Wprob, LL * DD * DD);
  cvt_kernel<<<4096, 256, 0, stream>>>(W1,    W1b,   LL * 4 * DD * DD);
  cvt_kernel<<<4096, 256, 0, stream>>>(W2,    W2b,   LL * DD * 4 * DD);
  cvt_kernel<<<8192, 256, 0, stream>>>(tok,   tokb,  VV * DD);

  embed_kernel<<<MM, 256, 0, stream>>>(ids, tok, pos, x);

  for (int l = 0; l < LL; ++l) {
    ln_kernel<<<MM, 256, 0, stream>>>(x, ln1g + l * DD, ln1b + l * DD, hb);
    gemm_kernel<<<dim3(3 * DD / BN, MM / BM), 256, 0, stream>>>(
        hb, Wqkvb + (size_t)l * 3 * DD * DD, bqkv + l * 3 * DD, nullptr,
        nullptr, qkv, 3 * DD, DD, 0);
    attn_kernel<<<2 * HH * (TT / 16), 32, 0, stream>>>(qkv, att);
    gemm_kernel<<<dim3(DD / BN, MM / BM), 256, 0, stream>>>(
        att, Wprob + (size_t)l * DD * DD, bproj + l * DD, x, x, nullptr,
        DD, DD, 0);
    ln_kernel<<<MM, 256, 0, stream>>>(x, ln2g + l * DD, ln2b + l * DD, hb);
    gemm_kernel<<<dim3(4 * DD / BN, MM / BM), 256, 0, stream>>>(
        hb, W1b + (size_t)l * 4 * DD * DD, b1 + l * 4 * DD, nullptr, nullptr,
        mid, 4 * DD, DD, 1);
    gemm_kernel<<<dim3(DD / BN, MM / BM), 256, 0, stream>>>(
        mid, W2b + (size_t)l * DD * 4 * DD, b2 + l * DD, x, x, nullptr,
        DD, 4 * DD, 0);
  }

  ln_kernel<<<MM, 256, 0, stream>>>(x, lnfg, lnfb, hb);
  gemm_kernel<<<dim3(VV / BN, MM / BM), 256, 0, stream>>>(
      hb, tokb, nullptr, nullptr, logits, nullptr, VV, DD, 0);
}
